// ResiSC_Decoder_69312182223717
// MI455X (gfx1250) — compile-verified, matrix operand order checked by reading, not run on hardware
//
#include <hip/hip_runtime.h>
#include <math.h>

// ---------------------------------------------------------------------------
// ResiSC decoder: 2-layer hetero-GAT (3 relations) + BN + final projection.
// Dense GEMMs via v_wmma_f32_16x16x32_f16 (wave32), operands staged in f16 so
// each WMMA fragment = two GLOBAL_LOAD_B128 per lane; each wave computes a
// 16x64 output strip (4 N-tiles) reusing its A fragments 4x. Edge phase
// gathers f16 fs/fd rows with b128 loads (L2-resident) + f32 atomics for
// segment max/sum.
// ---------------------------------------------------------------------------

typedef __attribute__((ext_vector_type(16))) _Float16 v16h;
typedef __attribute__((ext_vector_type(8)))  float    v8f;

#define HID 64
#define NHEAD 4
#define FDIM 256          // H * HID
#define EPS_BN 1e-5f
#define ATT_SCALE 0.125f  // 1/sqrt(64)

union Frag16 { v16h v; _Float16 h[16]; float4 q[2]; };
union Acc8   { v8f  v; float    f[8]; };
union H8     { float4 q; _Float16 h[8]; };

// ---------------------------------------------------------------------------
// GEMM: C[nrows,ncols] = act(A16 * B + bias), K = 64, ncols % 64 == 0.
// A16: [nrows,64] f16 row-major.  BT16: [ncols,64] f16 (B transposed).
// Per-lane fragment = two contiguous 16-byte chunks (ISA 16-bit A/B layout):
//   element e in 0..7  -> K = ks + 8*half + e
//   element e in 8..15 -> K = ks + 16 + 8*half + (e-8)
// One wave computes 4 adjacent 16x16 tiles (16x64 strip), reusing A frags.
// ---------------------------------------------------------------------------
template <int RELU, int HASBIAS, typename OutT>
__global__ void gemm64_wmma(const _Float16* __restrict__ A16,
                            const _Float16* __restrict__ BT16,
                            const float* __restrict__ bias,
                            OutT* __restrict__ C,
                            int nrows, int ncols) {
  const int groupsN = ncols >> 6;                 // groups of 4 n-tiles
  const int total = (nrows >> 4) * groupsN;
  const int t = blockIdx.x * (blockDim.x >> 5) + (threadIdx.x >> 5);
  if (t >= total) return;                         // wave-uniform: EXEC full
  const int lane = threadIdx.x & 31;
  const int half = (lane >> 4) & 1;
  const int tm = t / groupsN, tg = t % groupsN;
  const int mrow = (tm << 4) + (lane & 15);

  // A fragments for both k-steps (K=0..31 and K=32..63)
  const _Float16* arow = A16 + (size_t)mrow * HID + 8 * half;
  Frag16 a0, a1;
  a0.q[0] = *reinterpret_cast<const float4*>(arow);
  a0.q[1] = *reinterpret_cast<const float4*>(arow + 16);
  a1.q[0] = *reinterpret_cast<const float4*>(arow + 32);
  a1.q[1] = *reinterpret_cast<const float4*>(arow + 48);

  Acc8 acc[4];
#pragma unroll
  for (int j = 0; j < 4; ++j) {
    const int ncol = (tg << 6) + (j << 4) + (lane & 15);
    const _Float16* brow = BT16 + (size_t)ncol * HID + 8 * half;
    Frag16 b0, b1;
    b0.q[0] = *reinterpret_cast<const float4*>(brow);
    b0.q[1] = *reinterpret_cast<const float4*>(brow + 16);
    b1.q[0] = *reinterpret_cast<const float4*>(brow + 32);
    b1.q[1] = *reinterpret_cast<const float4*>(brow + 48);
    v8f z = {0.f, 0.f, 0.f, 0.f, 0.f, 0.f, 0.f, 0.f};
    acc[j].v = __builtin_amdgcn_wmma_f32_16x16x32_f16(
        false, a0.v, false, b0.v, (short)0, z, false, false);
    acc[j].v = __builtin_amdgcn_wmma_f32_16x16x32_f16(
        false, a1.v, false, b1.v, (short)0, acc[j].v, false, false);
  }

  // Epilogue: D layout -> M = vgpr + 8*half, N = lane&15 within tile.
#pragma unroll
  for (int j = 0; j < 4; ++j) {
    const int ncol = (tg << 6) + (j << 4) + (lane & 15);
    const float bv = HASBIAS ? bias[ncol] : 0.0f;
    OutT* crow = C + (size_t)((tm << 4) + 8 * half) * ncols + ncol;
#pragma unroll
    for (int r = 0; r < 8; ++r) {
      float v = acc[j].f[r] + bv;
      if (RELU) v = fmaxf(v, 0.0f);
      crow[(size_t)r * ncols] = (OutT)v;
    }
  }
}

// ---------------------------------------------------------------------------
__global__ void fill_f32(float* __restrict__ p, size_t n, float val) {
  size_t i = (size_t)blockIdx.x * blockDim.x + threadIdx.x;
  size_t stride = (size_t)gridDim.x * blockDim.x;
  for (; i < n; i += stride) p[i] = val;
}

__global__ void cvt_f32_to_f16(const float* __restrict__ src,
                               _Float16* __restrict__ dst, size_t n) {
  size_t i = (size_t)blockIdx.x * blockDim.x + threadIdx.x;
  size_t stride = (size_t)gridDim.x * blockDim.x;
  for (; i < n; i += stride) dst[i] = (_Float16)src[i];
}

// W: [nmats][64][ncols] f32 -> WT: [nmats][ncols][64] f16 (transposed)
__global__ void transpose_w_f16(const float* __restrict__ W,
                                _Float16* __restrict__ WT,
                                int ncols, int nmats) {
  const size_t total = (size_t)nmats * HID * ncols;
  size_t i = (size_t)blockIdx.x * blockDim.x + threadIdx.x;
  size_t stride = (size_t)gridDim.x * blockDim.x;
  for (; i < total; i += stride) {
    const int mat = (int)(i / ((size_t)HID * ncols));
    const int rem = (int)(i - (size_t)mat * HID * ncols);
    const int n = rem / HID;       // output row (= B column)
    const int k = rem - n * HID;   // output col (= B row)
    WT[i] = (_Float16)W[(size_t)mat * HID * ncols + (size_t)k * ncols + n];
  }
}

// head-mean: agg [N,4,64] f32 -> h16 [N,64] f16
__global__ void headmean_f16(const float* __restrict__ agg,
                             _Float16* __restrict__ h16, int nrows) {
  const size_t i = (size_t)blockIdx.x * blockDim.x + threadIdx.x;
  if (i >= (size_t)nrows * HID) return;
  const size_t n = i >> 6;
  const int d = (int)(i & (HID - 1));
  const float* p = agg + n * FDIM + d;
  h16[i] = (_Float16)(0.25f * (p[0] + p[64] + p[128] + p[192]));
}

__device__ __forceinline__ void atomicMaxFloat(float* addr, float val) {
  unsigned int* ai = (unsigned int*)addr;
  unsigned int old = *ai;
  while (true) {
    float of = __uint_as_float(old);
    if (of >= val) break;
    unsigned int assumed = old;
    old = atomicCAS(ai, assumed, __float_as_uint(val));
    if (old == assumed) break;
  }
}

// ---------------------------------------------------------------------------
// Pass 1: per-edge per-head scaled dot + segment max. One wave / edge.
// Lane L loads 8 contiguous f16 channels [8L, 8L+8) of fs[src] and fd[dst]
// (one b128 each); those 8 channels lie within head L>>3. Width-8 shuffle
// reduction leaves head sums on lanes 0,8,16,24.
// ---------------------------------------------------------------------------
__global__ void edge_scores(const _Float16* __restrict__ fs,
                            const _Float16* __restrict__ fd,
                            const int* __restrict__ src,
                            const int* __restrict__ dst,
                            float* __restrict__ e_out,
                            float* __restrict__ mmax, int E) {
  const int wid = blockIdx.x * (blockDim.x >> 5) + (threadIdx.x >> 5);
  if (wid >= E) return;
  const int lane = threadIdx.x & 31;
  const int s = src[wid], d = dst[wid];
  H8 a, b;
  a.q = *reinterpret_cast<const float4*>(fs + (size_t)s * FDIM + lane * 8);
  b.q = *reinterpret_cast<const float4*>(fd + (size_t)d * FDIM + lane * 8);
  float part = 0.0f;
#pragma unroll
  for (int j = 0; j < 8; ++j) part += (float)a.h[j] * (float)b.h[j];
#pragma unroll
  for (int off = 4; off > 0; off >>= 1)
    part += __shfl_down(part, off, 8);
  if ((lane & 7) == 0) {
    const int h = lane >> 3;
    const float ev = part * ATT_SCALE;
    e_out[(size_t)wid * NHEAD + h] = ev;
    atomicMaxFloat(&mmax[(size_t)d * NHEAD + h], ev);
  }
}

// Pass 2: p = exp(e - m[dst]); denom[dst] += p. One thread / (edge,head).
__global__ void edge_exp(float* __restrict__ e_buf,
                         const float* __restrict__ mmax,
                         const int* __restrict__ dst,
                         float* __restrict__ denom, int E) {
  const size_t i = (size_t)blockIdx.x * blockDim.x + threadIdx.x;
  if (i >= (size_t)E * NHEAD) return;
  const int eidx = (int)(i >> 2);
  const int h    = (int)(i & 3);
  const int d = dst[eidx];
  const float p = __expf(e_buf[i] - mmax[(size_t)d * NHEAD + h]);
  e_buf[i] = p;
  atomicAdd(&denom[(size_t)d * NHEAD + h], p);
}

// Pass 3: agg[dst] += fs[src] * alpha. One wave / edge; lane handles 8
// contiguous channels (one b128 gather, head uniform per lane).
__global__ void edge_aggregate(const _Float16* __restrict__ fs,
                               const float* __restrict__ p_buf,
                               const float* __restrict__ denom,
                               const int* __restrict__ src,
                               const int* __restrict__ dst,
                               float* __restrict__ agg, int E) {
  const int wid = blockIdx.x * (blockDim.x >> 5) + (threadIdx.x >> 5);
  if (wid >= E) return;
  const int lane = threadIdx.x & 31;
  const int s = src[wid], d = dst[wid];
  const int h = lane >> 3;
  const float alpha = p_buf[(size_t)wid * NHEAD + h] /
                      fmaxf(denom[(size_t)d * NHEAD + h], 1e-9f);
  H8 a;
  a.q = *reinterpret_cast<const float4*>(fs + (size_t)s * FDIM + lane * 8);
  float* pa = agg + (size_t)d * FDIM + lane * 8;
#pragma unroll
  for (int j = 0; j < 8; ++j)
    atomicAdd(&pa[j], (float)a.h[j] * alpha);
}

// ---------------------------------------------------------------------------
// BatchNorm stats: per-channel sum / sumsq over N rows.
__global__ void bn_partial(const float* __restrict__ x,
                           float* __restrict__ stat, int nrows) {
  const int c = threadIdx.x;            // blockDim.x == 64
  const int rows_per_block = (nrows + gridDim.x - 1) / gridDim.x;
  const int r0 = blockIdx.x * rows_per_block;
  const int r1 = min(r0 + rows_per_block, nrows);
  float s = 0.0f, s2 = 0.0f;
  for (int r = r0; r < r1; ++r) {
    const float v = x[(size_t)r * HID + c];
    s += v; s2 += v * v;
  }
  atomicAdd(&stat[c], s);
  atomicAdd(&stat[HID + c], s2);
}

// Normalize and write f16 (feeds next layer's WMMA GEMMs directly).
__global__ void bn_apply_f16(const float* __restrict__ x,
                             const float* __restrict__ stat,
                             const float* __restrict__ gamma,
                             const float* __restrict__ beta,
                             _Float16* __restrict__ y, int nrows) {
  const size_t i = (size_t)blockIdx.x * blockDim.x + threadIdx.x;
  if (i >= (size_t)nrows * HID) return;
  const int c = (int)(i & (HID - 1));
  const float invN = 1.0f / (float)nrows;
  const float mu = stat[c] * invN;
  const float var = stat[HID + c] * invN - mu * mu;
  y[i] = (_Float16)(gamma[c] * (x[i] - mu) * rsqrtf(var + EPS_BN) + beta[c]);
}

// ---------------------------------------------------------------------------
static inline size_t align256(size_t x) { return (x + 255) & ~(size_t)255; }

extern "C" void kernel_launch(void* const* d_in, const int* in_sizes, int n_in,
                              void* d_out, int out_size, void* d_ws, size_t ws_size,
                              hipStream_t stream) {
  const float* x     = (const float*)d_in[0];
  const int*   srcs[3] = {(const int*)d_in[1], (const int*)d_in[3], (const int*)d_in[5]};
  const int*   dsts[3] = {(const int*)d_in[2], (const int*)d_in[4], (const int*)d_in[6]};
  const float* Wsrc  = (const float*)d_in[7];   // [2,3,64,256]
  const float* Wdst  = (const float*)d_in[8];   // [2,3,64,256]
  const float* Wfc   = (const float*)d_in[9];   // [2,64,64]
  const float* bfc   = (const float*)d_in[10];  // [2,64]
  const float* gamma = (const float*)d_in[11];  // [2,64]
  const float* beta  = (const float*)d_in[12];  // [2,64]
  const float* Wdim  = (const float*)d_in[13];  // [64,256]
  const float* bdim  = (const float*)d_in[14];  // [256]
  float* out = (float*)d_out;

  const int N = in_sizes[0] / HID;   // 50000
  const int E = in_sizes[1];         // 250000

  // workspace layout
  char* ws = (char*)d_ws;
  size_t off = 0;
  float*    h_pre  = (float*)(ws + off);    off += align256((size_t)N * HID  * 4);
  float*    agg    = (float*)(ws + off);    off += align256((size_t)N * FDIM * 4);
  float*    e_buf  = (float*)(ws + off);    off += align256((size_t)E * NHEAD * 4);
  float*    mmax   = (float*)(ws + off);    off += align256((size_t)N * NHEAD * 4);
  float*    denom  = (float*)(ws + off);    off += align256((size_t)N * NHEAD * 4);
  float*    stat   = (float*)(ws + off);    off += align256(2 * HID * 4);
  _Float16* fs     = (_Float16*)(ws + off); off += align256((size_t)N * FDIM * 2);
  _Float16* fd     = (_Float16*)(ws + off); off += align256((size_t)N * FDIM * 2);
  _Float16* h16    = (_Float16*)(ws + off); off += align256((size_t)N * HID * 2);
  _Float16* wsrc16 = (_Float16*)(ws + off); off += align256((size_t)6 * HID * FDIM * 2);
  _Float16* wdst16 = (_Float16*)(ws + off); off += align256((size_t)6 * HID * FDIM * 2);
  _Float16* wfc16  = (_Float16*)(ws + off); off += align256((size_t)2 * HID * HID * 2);
  _Float16* wdim16 = (_Float16*)(ws + off); off += align256((size_t)HID * FDIM * 2);
  (void)ws_size; (void)n_in; (void)out_size;

  // ---- one-time (per launch) weight transpose+convert to f16 -------------
  transpose_w_f16<<<(int)(((size_t)6 * HID * FDIM + 255) / 256), 256, 0, stream>>>(
      Wsrc, wsrc16, FDIM, 6);
  transpose_w_f16<<<(int)(((size_t)6 * HID * FDIM + 255) / 256), 256, 0, stream>>>(
      Wdst, wdst16, FDIM, 6);
  transpose_w_f16<<<(int)(((size_t)2 * HID * HID + 255) / 256), 256, 0, stream>>>(
      Wfc, wfc16, HID, 2);
  transpose_w_f16<<<(int)(((size_t)HID * FDIM + 255) / 256), 256, 0, stream>>>(
      Wdim, wdim16, FDIM, 1);

  // h16 = f16(x)
  cvt_f32_to_f16<<<(int)(((size_t)N * HID + 255) / 256), 256, 0, stream>>>(
      x, h16, (size_t)N * HID);

  const int waveProj = (N / 16) * (FDIM / 64);         // 16x64 strips
  const int waveFc   = (N / 16) * (HID / 64);
  const dim3 gemmBlk(128);                             // 4 waves / block
  const int gridProj = (waveProj + 3) / 4;
  const int gridFc   = (waveFc + 3) / 4;
  const int wavesBlk = 256 / 32;                       // edge kernels: 8 edges/block
  const int gridEdgeW = (E + wavesBlk - 1) / wavesBlk;
  const int gridEdgeT = (int)(((size_t)E * NHEAD + 255) / 256);
  const int gridFillAgg = (int)(((size_t)N * FDIM + 255) / 256);
  const int gridFillNH  = (int)(((size_t)N * NHEAD + 255) / 256);
  const int gridNodeC   = (int)(((size_t)N * HID + 255) / 256);

  for (int l = 0; l < 2; ++l) {
    fill_f32<<<gridFillAgg, 256, 0, stream>>>(agg, (size_t)N * FDIM, 0.0f);

    for (int r = 0; r < 3; ++r) {
      const _Float16* ws_w = wsrc16 + (size_t)(l * 3 + r) * HID * FDIM;
      const _Float16* wd_w = wdst16 + (size_t)(l * 3 + r) * HID * FDIM;
      gemm64_wmma<0, 0, _Float16><<<gridProj, gemmBlk, 0, stream>>>(
          h16, ws_w, nullptr, fs, N, FDIM);
      gemm64_wmma<0, 0, _Float16><<<gridProj, gemmBlk, 0, stream>>>(
          h16, wd_w, nullptr, fd, N, FDIM);

      fill_f32<<<gridFillNH, 256, 0, stream>>>(mmax, (size_t)N * NHEAD, -3.0e38f);
      fill_f32<<<gridFillNH, 256, 0, stream>>>(denom, (size_t)N * NHEAD, 0.0f);

      edge_scores<<<gridEdgeW, 256, 0, stream>>>(fs, fd, srcs[r], dsts[r],
                                                 e_buf, mmax, E);
      edge_exp<<<gridEdgeT, 256, 0, stream>>>(e_buf, mmax, dsts[r], denom, E);
      edge_aggregate<<<gridEdgeW, 256, 0, stream>>>(fs, e_buf, denom,
                                                    srcs[r], dsts[r], agg, E);
    }

    // head-mean (agg -> h16), then fc GEMM (bias + relu) -> h_pre (f32)
    headmean_f16<<<gridNodeC, 256, 0, stream>>>(agg, h16, N);
    gemm64_wmma<1, 1, float><<<gridFc, gemmBlk, 0, stream>>>(
        h16, wfc16 + (size_t)l * HID * HID, bfc + (size_t)l * HID, h_pre, N, HID);

    // batchnorm -> h16 (f16, feeds next layer / final projection)
    fill_f32<<<1, 128, 0, stream>>>(stat, 2 * HID, 0.0f);
    bn_partial<<<128, HID, 0, stream>>>(h_pre, stat, N);
    bn_apply_f16<<<gridNodeC, 256, 0, stream>>>(h_pre, stat,
                                                gamma + (size_t)l * HID,
                                                beta + (size_t)l * HID, h16, N);
  }

  // final projection -> d_out [N, 256] f32
  gemm64_wmma<0, 1, float><<<gridProj, gemmBlk, 0, stream>>>(
      h16, wdim16, bdim, out, N, FDIM);
}
